// DSTAGNN_block_55920474193943
// MI455X (gfx1250) — compile-verified, matrix operand order checked by reading, not run on hardware
//
#include <hip/hip_runtime.h>
#include <hip/hip_bf16.h>

// ---------------- problem constants ----------------
#define B_ 16
#define N_ 1024
#define T_ 12
#define K_ 3
#define C_ 64
#define DM_ 512
#define DK_ 32
#define H_ 3
#define HD_ 96   // H*DK

typedef __attribute__((ext_vector_type(16))) _Float16 v16h;
typedef __attribute__((ext_vector_type(8)))  float    v8f;

union Frag16 { int4 i4[2]; v16h h; _Float16 e[16]; };
union F8     { float4 f4[2]; float f[8]; };
union H8     { int4 i4; _Float16 e[8]; };

static __device__ __forceinline__ v8f wmma_f16(v16h a, v16h b, v8f c) {
  return __builtin_amdgcn_wmma_f32_16x16x32_f16(false, a, false, b, (short)0, c, false, false);
}

// block-wide sum over 256 threads (8 waves, wave32)
static __device__ __forceinline__ float block_sum(float v, float* red8) {
#pragma unroll
  for (int o = 16; o > 0; o >>= 1) v += __shfl_xor(v, o, 32);
  __syncthreads();
  if ((threadIdx.x & 31) == 0) red8[threadIdx.x >> 5] = v;
  __syncthreads();
  float s = 0.f;
#pragma unroll
  for (int i = 0; i < 8; ++i) s += red8[i];
  return s;
}

// ===== kernel 0a: transposed bias/cheb:  biasT[k,m,n] = adj[n,m]*mask[k,n,m]; chebT[k,m,n]=cheb[k,n,m]
__global__ __launch_bounds__(256) void k_prep_trans(const float* __restrict__ adj,
                                                    const float* __restrict__ mask,
                                                    const float* __restrict__ cheb,
                                                    float* __restrict__ biasT,
                                                    float* __restrict__ chebT) {
  size_t i = (size_t)blockIdx.x * 256 + threadIdx.x;
  if (i >= (size_t)K_ * N_ * N_) return;
  int k = (int)(i / ((size_t)N_ * N_));
  int r = (int)(i % ((size_t)N_ * N_));
  int m = r / N_, n = r % N_;
  size_t src = (size_t)k * N_ * N_ + (size_t)n * N_ + m;
  biasT[i] = adj[(size_t)n * N_ + m] * mask[src];
  chebT[i] = cheb[src];
}

// ===== kernel 0b: pre-swizzle Wq_s/Wk_s into per-lane fragment layout (f16)
// layout: (((mat*16+ks)*6+jt)*32+lane)*16 + e
__global__ __launch_bounds__(256) void k_swz_qk(const float* __restrict__ Wq,
                                                const float* __restrict__ Wk,
                                                _Float16* __restrict__ out) {
  int idx = blockIdx.x * 256 + threadIdx.x;      // < 98304
  int e = idx & 15, lane = (idx >> 4) & 31, r = idx >> 9;
  int jt = r % 6, ks = (r / 6) % 16, mat = r / 96;
  int hs = lane >> 4, lr = lane & 15;
  int kk = ks * 32 + ((e < 8) ? (hs * 8 + e) : (16 + hs * 8 + (e - 8)));
  const float* W = mat ? Wk : Wq;
  out[idx] = (_Float16)W[(size_t)kk * HD_ + jt * 16 + lr];
}

// ===== kernel 0c: pre-swizzle conv weights (f16), layout (((ks*8)+jtt)*32+lane)*16+e
template <int KW>
__global__ __launch_bounds__(256) void k_swz_conv(const float* __restrict__ wg,
                                                  _Float16* __restrict__ out) {
  int idx = blockIdx.x * 256 + threadIdx.x;      // < 2*KW*4096
  int e = idx & 15, lane = (idx >> 4) & 31, jtt = (idx >> 9) & 7, ks = idx >> 12;
  int hs = lane >> 4, lr = lane & 15;
  int kk = ks * 32 + ((e < 8) ? (hs * 8 + e) : (16 + hs * 8 + (e - 8)));
  int ic = kk / KW, dt = kk % KW, oc = jtt * 16 + lr;
  out[idx] = (_Float16)wg[((size_t)oc * C_ + ic) * KW + dt];
}

// ================= kernel 1: TEmx = LN_n(x[b,:,0,t] + pos_T[t,:]) =================
__global__ __launch_bounds__(256) void k_temporal_embed(const float* __restrict__ x,
                                                        const float* __restrict__ pos_T,
                                                        const float* __restrict__ g,
                                                        const float* __restrict__ bb,
                                                        float* __restrict__ TEmx) {
  __shared__ float red8[8];
  int bt = blockIdx.x, b = bt / T_, t = bt % T_;
  float v[4], s = 0.f, ss = 0.f;
#pragma unroll
  for (int i = 0; i < 4; ++i) {
    int n = threadIdx.x + i * 256;
    float val = x[((size_t)b * N_ + n) * T_ + t] + pos_T[t * N_ + n];
    v[i] = val; s += val; ss += val * val;
  }
  s  = block_sum(s,  red8);
  ss = block_sum(ss, red8);
  float mean = s * (1.f / N_);
  float inv  = rsqrtf(ss * (1.f / N_) - mean * mean + 1e-5f);
#pragma unroll
  for (int i = 0; i < 4; ++i) {
    int n = threadIdx.x + i * 256;
    TEmx[((size_t)b * T_ + t) * N_ + n] = (v[i] - mean) * inv * g[n] + bb[n];
  }
}

// ================= kernel 2: temporal MHA + LN =================
__global__ __launch_bounds__(256) void k_temporal_attn(
    const float* __restrict__ TEmx, const float* __restrict__ res_att,
    const float* __restrict__ Wq, const float* __restrict__ Wk, const float* __restrict__ Wv,
    const float* __restrict__ Wfc_t, const float* __restrict__ ln_g, const float* __restrict__ ln_b,
    float* __restrict__ TAT, float* __restrict__ out_scores) {
  int b = blockIdx.x;
  __shared__ float qkv[3][T_][HD_];
  __shared__ float sc[H_][T_][T_];
  __shared__ float attn[H_][T_][T_];
  __shared__ float ctx[T_][HD_];
  __shared__ float red8[8];
  const float* te = TEmx + (size_t)b * T_ * N_;

  for (int idx = threadIdx.x; idx < 3 * T_ * HD_; idx += 256) {
    int which = idx / (T_ * HD_), rem = idx % (T_ * HD_);
    int t = rem / HD_, e = rem % HD_;
    const float* W = (which == 0) ? Wq : (which == 1) ? Wk : Wv;
    float acc = 0.f;
    for (int n = 0; n < N_; ++n) acc += te[t * N_ + n] * W[n * HD_ + e];
    qkv[which][t][e] = acc;
  }
  __syncthreads();
  const float invsq = 0.17677669529663689f; // 1/sqrt(32)
  for (int idx = threadIdx.x; idx < H_ * T_ * T_; idx += 256) {
    int h = idx / (T_ * T_), r = idx % (T_ * T_), q = r / T_, kk = r % T_;
    float acc = 0.f;
#pragma unroll
    for (int d = 0; d < DK_; ++d) acc += qkv[0][q][h * DK_ + d] * qkv[1][kk][h * DK_ + d];
    acc = acc * invsq + res_att[(((size_t)b * H_ + h) * T_ + q) * T_ + kk];
    sc[h][q][kk] = acc;
    out_scores[(((size_t)b * H_ + h) * T_ + q) * T_ + kk] = acc;
  }
  __syncthreads();
  // softmax over QUERY axis (axis=3 in reference), i.e. over q for fixed (h,kk)
  for (int idx = threadIdx.x; idx < H_ * T_; idx += 256) {
    int h = idx / T_, kk = idx % T_;
    float mx = -3e38f;
    for (int q = 0; q < T_; ++q) mx = fmaxf(mx, sc[h][q][kk]);
    float s = 0.f;
    for (int q = 0; q < T_; ++q) s += __expf(sc[h][q][kk] - mx);
    float inv = 1.f / s;
    for (int q = 0; q < T_; ++q) attn[h][q][kk] = __expf(sc[h][q][kk] - mx) * inv;
  }
  __syncthreads();
  for (int idx = threadIdx.x; idx < T_ * HD_; idx += 256) {
    int t = idx / HD_, e = idx % HD_, h = e / DK_;
    float acc = 0.f;
#pragma unroll
    for (int kk = 0; kk < T_; ++kk) acc += attn[h][t][kk] * qkv[2][kk][e];
    ctx[t][e] = acc;
  }
  __syncthreads();
  // TATout = LN_n(ctx @ Wfc_t + TEmx): two-pass recompute
  for (int t = 0; t < T_; ++t) {
    float s = 0.f, ss = 0.f;
    for (int n = threadIdx.x; n < N_; n += 256) {
      float a = te[t * N_ + n];
#pragma unroll
      for (int e = 0; e < HD_; ++e) a += ctx[t][e] * Wfc_t[(size_t)e * N_ + n];
      s += a; ss += a * a;
    }
    s  = block_sum(s,  red8);
    ss = block_sum(ss, red8);
    float mean = s * (1.f / N_);
    float inv  = rsqrtf(ss * (1.f / N_) - mean * mean + 1e-5f);
    for (int n = threadIdx.x; n < N_; n += 256) {
      float a = te[t * N_ + n];
#pragma unroll
      for (int e = 0; e < HD_; ++e) a += ctx[t][e] * Wfc_t[(size_t)e * N_ + n];
      TAT[((size_t)b * T_ + t) * N_ + n] = (a - mean) * inv * ln_g[n] + ln_b[n];
    }
  }
}

// ================= kernel 3: x_TAt -> LN -> SEmx(f16) -> WMMA Q/K projections =================
__global__ __launch_bounds__(256) void k_spatial_qk(
    const float* __restrict__ TAT, const float* __restrict__ Wpre, const float* __restrict__ bpre,
    const float* __restrict__ pos_S, const float* __restrict__ g, const float* __restrict__ bb,
    const _Float16* __restrict__ Wqksw,
    _Float16* __restrict__ Qh, _Float16* __restrict__ Kh) {
  int b = blockIdx.x >> 6;
  int n0 = (blockIdx.x & 63) * 16;
  __shared__ float A32[16][DM_];
  __shared__ alignas(16) _Float16 A16[16][DM_];
  __shared__ float red8[8];

  for (int idx = threadIdx.x; idx < 16 * DM_; idx += 256) {
    int r = idx >> 9, d = idx & (DM_ - 1);
    int n = n0 + r;
    float acc = bpre[d];
#pragma unroll
    for (int t = 0; t < T_; ++t) acc += TAT[((size_t)b * T_ + t) * N_ + n] * Wpre[t * DM_ + d];
    A32[r][d] = acc + pos_S[(size_t)n * DM_ + d];
  }
  __syncthreads();
  for (int r = 0; r < 16; ++r) {
    float s = 0.f, ss = 0.f;
    for (int d = threadIdx.x; d < DM_; d += 256) { float v = A32[r][d]; s += v; ss += v * v; }
    s  = block_sum(s,  red8);
    ss = block_sum(ss, red8);
    float mean = s * (1.f / DM_);
    float inv  = rsqrtf(ss * (1.f / DM_) - mean * mean + 1e-5f);
    for (int d = threadIdx.x; d < DM_; d += 256)
      A16[r][d] = (_Float16)((A32[r][d] - mean) * inv * g[d] + bb[d]);
  }
  __syncthreads();

  int wave = threadIdx.x >> 5, lane = threadIdx.x & 31;
  int lr = lane & 15, hs = lane >> 4;
  for (int job = wave; job < 12; job += 8) {      // 2 mats x 6 col tiles of 16
    int mat = job / 6, jt = job % 6, e0 = jt * 16;
    v8f acc = {};
    for (int ks = 0; ks < 16; ++ks) {
      int k0 = ks * 32;
      Frag16 ua, ub;
      ua.i4[0] = *(const int4*)&A16[lr][k0 + hs * 8];
      ua.i4[1] = *(const int4*)&A16[lr][k0 + 16 + hs * 8];
      const _Float16* wp = Wqksw + ((((size_t)mat * 16 + ks) * 6 + jt) * 32 + lane) * 16;
      ub.i4[0] = *(const int4*)wp;
      ub.i4[1] = *(const int4*)(wp + 8);
      acc = wmma_f16(ua.h, ub.h, acc);
    }
    _Float16* dst = mat ? Kh : Qh;
#pragma unroll
    for (int v = 0; v < 8; ++v) {
      int row = v + hs * 8, col = lr;
      int e = e0 + col, kk = e >> 5, dk = e & 31;
      dst[(((size_t)b * K_ + kk) * N_ + n0 + row) * DK_ + dk] = (_Float16)acc[v];
    }
  }
}

// ===== kernel 4: fused STAt WMMA + column softmax + cheb conv + Theta + relu =====
__global__ __launch_bounds__(256) void k_att_cheb(
    const _Float16* __restrict__ Qh, const _Float16* __restrict__ Kh,
    const float* __restrict__ biasT, const float* __restrict__ chebT,
    const float* __restrict__ x, const float* __restrict__ Theta,
    _Float16* __restrict__ sg) {
  int b  = blockIdx.x >> 3;
  int m0 = (blockIdx.x & 7) * 128;
  __shared__ float wred[8][128];
  __shared__ float colmax[128];
  __shared__ float colsum[128];
  __shared__ float rhs[K_][128][T_];
  __shared__ alignas(16) _Float16 Wtile[128][128];   // [m][n]
  __shared__ alignas(16) float xt[128 * T_];
  int wave = threadIdx.x >> 5, lane = threadIdx.x & 31;
  int lr = lane & 15, hs = lane >> 4;
  const float invsq = 0.17677669529663689f;

  for (int k = 0; k < K_; ++k) {
    const _Float16* Qb = Qh + ((size_t)b * K_ + k) * N_ * DK_;
    const _Float16* Kb = Kh + ((size_t)b * K_ + k) * N_ * DK_;
    const float* bsT = biasT + (size_t)k * N_ * N_;
    const float* chT = chebT + (size_t)k * N_ * N_;

    // -------- pass 1: column max --------
    for (int m = lane; m < 128; m += 32) wred[wave][m] = -3e38f;
    __syncthreads();
    for (int nt = 0; nt < 8; ++nt) {
      int n0 = nt * 128;
      int nrow = n0 + wave * 16 + lr;
      int nbase = n0 + wave * 16 + hs * 8;
      const _Float16* qr = Qb + (size_t)nrow * DK_;
      if (nt < 7) __builtin_prefetch(qr + 128 * DK_, 0, 2);
      Frag16 ua;
      ua.i4[0] = *(const int4*)(qr + hs * 8);
      ua.i4[1] = *(const int4*)(qr + 16 + hs * 8);
      for (int j = 0; j < 8; ++j) {
        int cc = m0 + j * 16 + lr;
        const _Float16* kr = Kb + (size_t)cc * DK_;
        Frag16 ub;
        ub.i4[0] = *(const int4*)(kr + hs * 8);
        ub.i4[1] = *(const int4*)(kr + 16 + hs * 8);
        v8f s = {};
        s = wmma_f16(ua.h, ub.h, s);
        F8 bi;
        const float* bcol = bsT + (size_t)cc * N_ + nbase;
        bi.f4[0] = *(const float4*)bcol;
        bi.f4[1] = *(const float4*)(bcol + 4);
        float mx = -3e38f;
#pragma unroll
        for (int v = 0; v < 8; ++v) mx = fmaxf(mx, s[v] * invsq + bi.f[v]);
        mx = fmaxf(mx, __shfl_xor(mx, 16, 32));
        if (lane < 16) { int ml = j * 16 + lane; wred[wave][ml] = fmaxf(wred[wave][ml], mx); }
      }
    }
    __syncthreads();
    if (threadIdx.x < 128) {
      float mx = wred[0][threadIdx.x];
#pragma unroll
      for (int w = 1; w < 8; ++w) mx = fmaxf(mx, wred[w][threadIdx.x]);
      colmax[threadIdx.x] = mx;
    }
    __syncthreads();
    if (threadIdx.x < 128) colsum[threadIdx.x] = 0.f;
    for (int idx = threadIdx.x; idx < 128 * T_; idx += 256) rhs[k][idx / T_][idx % T_] = 0.f;
    for (int m = lane; m < 128; m += 32) wred[wave][m] = 0.f;

    // -------- pass 2: exp, cheb weight, rhs accumulation --------
    for (int nt = 0; nt < 8; ++nt) {
      int n0 = nt * 128;
      __syncthreads(); // previous rhs accumulation done; Wtile/xt free
      {
        const float4* xsrc = (const float4*)(x + ((size_t)b * N_ + n0) * T_);
        float4* xdst = (float4*)xt;
        for (int i = threadIdx.x; i < (128 * T_) / 4; i += 256) xdst[i] = xsrc[i];
      }
      int nrow = n0 + wave * 16 + lr;
      int nbase = n0 + wave * 16 + hs * 8;
      const _Float16* qr = Qb + (size_t)nrow * DK_;
      Frag16 ua;
      ua.i4[0] = *(const int4*)(qr + hs * 8);
      ua.i4[1] = *(const int4*)(qr + 16 + hs * 8);
      for (int j = 0; j < 8; ++j) {
        int cc = m0 + j * 16 + lr;
        const _Float16* kr = Kb + (size_t)cc * DK_;
        Frag16 ub;
        ub.i4[0] = *(const int4*)(kr + hs * 8);
        ub.i4[1] = *(const int4*)(kr + 16 + hs * 8);
        v8f s = {};
        s = wmma_f16(ua.h, ub.h, s);
        F8 bi, ch;
        const float* bcol = bsT + (size_t)cc * N_ + nbase;
        const float* ccol = chT + (size_t)cc * N_ + nbase;
        bi.f4[0] = *(const float4*)bcol;  bi.f4[1] = *(const float4*)(bcol + 4);
        ch.f4[0] = *(const float4*)ccol;  ch.f4[1] = *(const float4*)(ccol + 4);
        float cm = colmax[j * 16 + lr];
        float psum = 0.f;
        H8 wv;
#pragma unroll
        for (int v = 0; v < 8; ++v) {
          float ev = __expf(s[v] * invsq + bi.f[v] - cm);
          psum += ev;
          wv.e[v] = (_Float16)(ev * ch.f[v]);
        }
        *(int4*)&Wtile[j * 16 + lr][wave * 16 + hs * 8] = wv.i4;
        psum += __shfl_xor(psum, 16, 32);
        if (lane < 16) wred[wave][j * 16 + lane] += psum;
      }
      __syncthreads(); // Wtile + xt ready
      for (int idx = threadIdx.x; idx < 128 * T_; idx += 256) {
        int m = idx / T_, t = idx % T_;
        float acc = 0.f;
        for (int n8 = 0; n8 < 16; ++n8) {
          H8 wv; wv.i4 = *(const int4*)&Wtile[m][n8 * 8];
#pragma unroll
          for (int u = 0; u < 8; ++u) acc += (float)wv.e[u] * xt[(n8 * 8 + u) * T_ + t];
        }
        rhs[k][m][t] += acc;
      }
    }
    __syncthreads();
    if (threadIdx.x < 128) {
      float s = 0.f;
#pragma unroll
      for (int w = 0; w < 8; ++w) s += wred[w][threadIdx.x];
      colsum[threadIdx.x] = s;
    }
    __syncthreads();
    for (int idx = threadIdx.x; idx < 128 * T_; idx += 256) {
      int m = idx / T_, t = idx % T_;
      rhs[k][m][t] /= colsum[m];
    }
    __syncthreads();
  }
  // epilogue: sg[b, m0+m, c, t] = relu(sum_k rhs_k[m,t]*Theta[k,0,c])  (f16)
  for (int idx = threadIdx.x; idx < 128 * C_ * T_; idx += 256) {
    int m = idx / (C_ * T_), r = idx % (C_ * T_), c = r / T_, t = r % T_;
    float acc = rhs[0][m][t] * Theta[c] + rhs[1][m][t] * Theta[C_ + c] + rhs[2][m][t] * Theta[2 * C_ + c];
    sg[(((size_t)b * N_ + m0 + m) * C_ + c) * T_ + t] = (_Float16)fmaxf(acc, 0.f);
  }
}

// ===== kernel 5: gated temporal convs (WMMA implicit GEMM) + Wfc + residual + LN =====
template <int KW, int LOUT>
static __device__ __forceinline__ void conv_gtu_stage(
    int b, int n0, const _Float16* __restrict__ sg,
    const _Float16* __restrict__ Wc, const float* __restrict__ bg,
    int loff, _Float16* __restrict__ gb, int wave, int lane) {
  constexpr int KSTEPS = 2 * KW;        // 64*KW/32
  int lr = lane & 15, hs = lane >> 4;
  const int njobs = 4 * LOUT;           // 4 gate-pairs (c-tiles of 16) x L positions
  for (int job = wave; job < njobs; job += 8) {
    int jt = job / LOUT, l = job % LOUT;
    v8f ylo = {}, yhi = {};
    const _Float16* srow = sg + ((size_t)b * N_ + n0 + lr) * C_ * T_;
    for (int ks = 0; ks < KSTEPS; ++ks) {
      int k0 = ks * 32;
      v16h a;
#pragma unroll
      for (int e = 0; e < 16; ++e) {
        int kk = k0 + ((e < 8) ? (hs * 8 + e) : (16 + hs * 8 + (e - 8)));
        int ic = kk / KW, dt = kk % KW;
        a[e] = srow[ic * T_ + l + dt];
      }
      Frag16 bl, bh;
      const _Float16* wl = Wc + (((size_t)ks * 8 + jt) * 32 + lane) * 16;
      const _Float16* wh = Wc + (((size_t)ks * 8 + jt + 4) * 32 + lane) * 16;
      bl.i4[0] = *(const int4*)wl;  bl.i4[1] = *(const int4*)(wl + 8);
      bh.i4[0] = *(const int4*)wh;  bh.i4[1] = *(const int4*)(wh + 8);
      ylo = wmma_f16(a, bl.h, ylo);
      yhi = wmma_f16(a, bh.h, yhi);
    }
    int c = jt * 16 + lr;
#pragma unroll
    for (int v = 0; v < 8; ++v) {
      int nr = hs * 8 + v;
      float gl = tanhf(ylo[v] + bg[c]);
      float gh = 1.f / (1.f + __expf(-(yhi[v] + bg[c + C_])));
      gb[((size_t)(nr * C_ + c)) * 24 + loff + l] = (_Float16)(gl * gh);
    }
  }
}

__global__ __launch_bounds__(256) void k_gtu_out(
    const _Float16* __restrict__ sg,
    const _Float16* __restrict__ Wc3, const float* __restrict__ bg3,
    const _Float16* __restrict__ Wc5, const float* __restrict__ bg5,
    const _Float16* __restrict__ Wc7, const float* __restrict__ bg7,
    const float* __restrict__ Wfc, const float* __restrict__ bfc,
    const float* __restrict__ x, const float* __restrict__ wres, const float* __restrict__ bres,
    const float* __restrict__ ln_g, const float* __restrict__ ln_b,
    float* __restrict__ out) {
  int b = blockIdx.x >> 6;
  int n0 = (blockIdx.x & 63) * 16;
  __shared__ alignas(16) _Float16 gbuf[16 * C_ * 24];   // 48 KB; later reused as float[16*C_*12]
  int wave = threadIdx.x >> 5, lane = threadIdx.x & 31;

  conv_gtu_stage<3, 10>(b, n0, sg, Wc3, bg3, 0,  gbuf, wave, lane);
  conv_gtu_stage<5, 8 >(b, n0, sg, Wc5, bg5, 10, gbuf, wave, lane);
  conv_gtu_stage<7, 6 >(b, n0, sg, Wc7, bg7, 18, gbuf, wave, lane);
  __syncthreads();

  // per-(n,c) block: consume 24 gated halves, produce 12 floats IN PLACE (same 48B region)
  float* ybuf = (float*)gbuf;
#pragma unroll
  for (int p = 0; p < 4; ++p) {
    int pc = threadIdx.x + p * 256;          // n*64 + c
    int n = pc >> 6, c = pc & 63;
    _Float16 lv[24];
#pragma unroll
    for (int l = 0; l < 24; ++l) lv[l] = gbuf[pc * 24 + l];
    const float* xrow = x + ((size_t)b * N_ + n0 + n) * T_;
    float w_res = wres[c], b_res = bres[c];
    for (int t = 0; t < T_; ++t) {
      float acc = bfc[t] + xrow[t] * w_res + b_res;
#pragma unroll
      for (int l = 0; l < 24; ++l) acc += (float)lv[l] * Wfc[l * T_ + t];
      ybuf[pc * T_ + t] = fmaxf(acc, 0.f);
    }
  }
  __syncthreads();

  // LN over C per (n,t); one wave per row, lanes cover c and c+32
  for (int row = wave; row < 16 * T_; row += 8) {
    int n = row / T_, t = row % T_;
    float a0 = ybuf[(n * C_ + lane) * T_ + t];
    float a1 = ybuf[(n * C_ + lane + 32) * T_ + t];
    float s = a0 + a1, ss = a0 * a0 + a1 * a1;
#pragma unroll
    for (int o = 16; o > 0; o >>= 1) { s += __shfl_xor(s, o, 32); ss += __shfl_xor(ss, o, 32); }
    float mean = s * (1.f / C_);
    float inv  = rsqrtf(ss * (1.f / C_) - mean * mean + 1e-5f);
    size_t base = (((size_t)b * N_ + n0 + n) * C_) * T_ + t;
    out[base + (size_t)lane * T_]        = (a0 - mean) * inv * ln_g[lane]      + ln_b[lane];
    out[base + (size_t)(lane + 32) * T_] = (a1 - mean) * inv * ln_g[lane + 32] + ln_b[lane + 32];
  }
}

// ================= host launcher =================
extern "C" void kernel_launch(void* const* d_in, const int* in_sizes, int n_in,
                              void* d_out, int out_size, void* d_ws, size_t ws_size,
                              hipStream_t stream) {
  (void)in_sizes; (void)n_in; (void)out_size; (void)ws_size;
  const float* x      = (const float*)d_in[0];
  const float* res_at = (const float*)d_in[1];
  const float* adj_pa = (const float*)d_in[2];
  const float* cheb   = (const float*)d_in[3];
  const float* pos_T  = (const float*)d_in[4];
  const float* gT_g   = (const float*)d_in[5];
  const float* gT_b   = (const float*)d_in[6];
  const float* Wq_t   = (const float*)d_in[7];
  const float* Wk_t   = (const float*)d_in[8];
  const float* Wv_t   = (const float*)d_in[9];
  const float* Wfc_t  = (const float*)d_in[10];
  const float* lnT_g  = (const float*)d_in[11];
  const float* lnT_b  = (const float*)d_in[12];
  const float* Wpre   = (const float*)d_in[13];
  const float* bpre   = (const float*)d_in[14];
  const float* pos_S  = (const float*)d_in[15];
  const float* gS_g   = (const float*)d_in[16];
  const float* gS_b   = (const float*)d_in[17];
  const float* Wq_s   = (const float*)d_in[18];
  const float* Wk_s   = (const float*)d_in[19];
  const float* Theta  = (const float*)d_in[20];
  const float* maskp  = (const float*)d_in[21];
  const float* wg3    = (const float*)d_in[22];
  const float* bg3    = (const float*)d_in[23];
  const float* wg5    = (const float*)d_in[24];
  const float* bg5    = (const float*)d_in[25];
  const float* wg7    = (const float*)d_in[26];
  const float* bg7    = (const float*)d_in[27];
  const float* wres   = (const float*)d_in[28];
  const float* bres   = (const float*)d_in[29];
  const float* Wfc    = (const float*)d_in[30];
  const float* bfc    = (const float*)d_in[31];
  const float* lnF_g  = (const float*)d_in[32];
  const float* lnF_b  = (const float*)d_in[33];

  char* ws = (char*)d_ws;
  float* TEmx  = (float*)ws;      ws += (size_t)B_ * T_ * N_ * 4;        // 786 KB
  float* TAT   = (float*)ws;      ws += (size_t)B_ * T_ * N_ * 4;        // 786 KB
  float* biasT = (float*)ws;      ws += (size_t)K_ * N_ * N_ * 4;        // 12.6 MB
  float* chebT = (float*)ws;      ws += (size_t)K_ * N_ * N_ * 4;        // 12.6 MB
  _Float16* Qh = (_Float16*)ws;   ws += (size_t)B_ * K_ * N_ * DK_ * 2;  // 3 MB
  _Float16* Kh = (_Float16*)ws;   ws += (size_t)B_ * K_ * N_ * DK_ * 2;  // 3 MB
  _Float16* sg = (_Float16*)ws;   ws += (size_t)B_ * N_ * C_ * T_ * 2;   // 25 MB
  _Float16* Wqksw = (_Float16*)ws; ws += (size_t)2 * 16 * 6 * 32 * 16 * 2;  // 192 KB
  _Float16* Wc3 = (_Float16*)ws;  ws += (size_t)6  * 8 * 32 * 16 * 2;    // 48 KB
  _Float16* Wc5 = (_Float16*)ws;  ws += (size_t)10 * 8 * 32 * 16 * 2;    // 80 KB
  _Float16* Wc7 = (_Float16*)ws;  ws += (size_t)14 * 8 * 32 * 16 * 2;    // 112 KB

  float* o_out = (float*)d_out;
  float* scores_out = o_out + (size_t)B_ * N_ * C_ * T_;

  k_prep_trans    <<<(K_ * N_ * N_) / 256, 256, 0, stream>>>(adj_pa, maskp, cheb, biasT, chebT);
  k_swz_qk        <<<98304 / 256, 256, 0, stream>>>(Wq_s, Wk_s, Wqksw);
  k_swz_conv<3>   <<<(6  * 4096) / 256, 256, 0, stream>>>(wg3, Wc3);
  k_swz_conv<5>   <<<(10 * 4096) / 256, 256, 0, stream>>>(wg5, Wc5);
  k_swz_conv<7>   <<<(14 * 4096) / 256, 256, 0, stream>>>(wg7, Wc7);
  k_temporal_embed<<<B_ * T_, 256, 0, stream>>>(x, pos_T, gT_g, gT_b, TEmx);
  k_temporal_attn <<<B_, 256, 0, stream>>>(TEmx, res_at, Wq_t, Wk_t, Wv_t, Wfc_t,
                                           lnT_g, lnT_b, TAT, scores_out);
  k_spatial_qk    <<<B_ * (N_ / 16), 256, 0, stream>>>(TAT, Wpre, bpre, pos_S, gS_g, gS_b,
                                                       Wqksw, Qh, Kh);
  k_att_cheb      <<<B_ * (N_ / 128), 256, 0, stream>>>(Qh, Kh, biasT, chebT, x, Theta, sg);
  k_gtu_out       <<<B_ * (N_ / 16), 256, 0, stream>>>(sg, Wc3, bg3, Wc5, bg5, Wc7, bg7,
                                                       Wfc, bfc, x, wres, bres, lnF_g, lnF_b, o_out);
}
